// Agent_70824010711254
// MI455X (gfx1250) — matrix operand order repaired and checked by reading, not compiled, not run on hardware
//
#include <hip/hip_runtime.h>
#include <math.h>

typedef __attribute__((ext_vector_type(16))) _Float16 v16h;
typedef __attribute__((ext_vector_type(8)))  _Float16 v8h;
typedef __attribute__((ext_vector_type(8)))  float    v8f;

#define OBS 17
#define HID 64
#define ACT 4
#define EPSQ 1e-4f
#define LOG2PI 1.8378770664093453f
#define WAVES 4
#define ROWS_PER_BLOCK (WAVES * 32)

// ---- fast activation: use CDNA5 hardware V_TANH_F32 when available ----
__device__ __forceinline__ float fast_tanh(float v) {
#if __has_builtin(__builtin_amdgcn_tanhf)
    return __builtin_amdgcn_tanhf(v);
#else
    // branch-free fallback: tanh(x) = sign(x) * (1 - e) / (1 + e), e = exp(-2|x|)
    float a = fabsf(v);
    float e = __expf(-2.f * a);
    float t = (1.f - e) * __builtin_amdgcn_rcpf(1.f + e);
    return copysignf(t, v);
#endif
}

// ---- WMMA fragment helpers (CDNA5 16x16x32 f16 layouts, CDNA5 ISA) ----

__device__ __forceinline__ v16h make16(v8h lo, v8h hi) {
    return __builtin_shufflevector(lo, hi, 0, 1, 2, 3, 4, 5, 6, 7,
                                           8, 9, 10, 11, 12, 13, 14, 15);
}

// A (16xK f16, row-major in LDS, row stride ld halves).
// Lane L: M = L%16, half = L/16. Element e -> K = k0 + e + 8*half (e<8)
//                                       K = k0 + e + 8 + 8*half (e>=8)
__device__ __forceinline__ v16h loadA(const _Float16* base, int ld, int m, int half, int k0) {
    const _Float16* p = base + m * ld + k0 + 8 * half;
    v8h lo = *(const v8h*)p;
    v8h hi = *(const v8h*)(p + 16);
    return make16(lo, hi);
}

// B (Kx16 f16) stored TRANSPOSED in LDS as Wt[n][k], row stride ld halves.
// Lane L: N = L%16, element e -> K = k0 + e + 8*(L/16) (e<8), K = k0 + e + 8 + 8*(L/16) (e>=8)
__device__ __forceinline__ v16h loadB(const _Float16* base, int ld, int n, int half, int k0) {
    const _Float16* p = base + n * ld + k0 + 8 * half;
    v8h lo = *(const v8h*)p;
    v8h hi = *(const v8h*)(p + 16);
    return make16(lo, hi);
}

__device__ __forceinline__ v8f wmma_f16(v16h a, v16h b, v8f c) {
    return __builtin_amdgcn_wmma_f32_16x16x32_f16(false, a, false, b,
                                                  (short)0, c, false, false);
}

// Hidden layer: out[32][HID] = tanh(in[32][KDIM] @ W + b). KDIM in {32, 64}.
template <int KDIM>
__device__ __forceinline__ void hidden_layer(const _Float16* in, int inLd,
                                             const _Float16* Wt,
                                             const float* bias,
                                             _Float16* out, int lane) {
    const int half = lane >> 4, l16 = lane & 15;
    constexpr int NK = KDIM / 32;
#pragma unroll
    for (int mt = 0; mt < 2; ++mt) {
        v16h a[NK];
#pragma unroll
        for (int c = 0; c < NK; ++c) a[c] = loadA(in, inLd, mt * 16 + l16, half, c * 32);
#pragma unroll
        for (int nt = 0; nt < 4; ++nt) {
            v8f acc = {0.f, 0.f, 0.f, 0.f, 0.f, 0.f, 0.f, 0.f};
#pragma unroll
            for (int c = 0; c < NK; ++c) {
                v16h b = loadB(Wt, KDIM, nt * 16 + l16, half, c * 32);
                acc = wmma_f16(a[c], b, acc);
            }
            const int ncol = nt * 16 + l16;
            const float bv = bias[ncol];
#pragma unroll
            for (int r = 0; r < 8; ++r) {
                int m = mt * 16 + r + 8 * half;
                out[m * HID + ncol] = (_Float16)fast_tanh(acc[r] + bv);
            }
        }
    }
}

// Final layer: outTile[32][ncols] (f32) = in[32][64] @ W3 + b (W3t padded to [16][64])
__device__ __forceinline__ void final_layer(const _Float16* in,
                                            const _Float16* W3t,
                                            float* outTile, int outLd, int ncols,
                                            const float* bias, int lane) {
    const int half = lane >> 4, l16 = lane & 15;
#pragma unroll
    for (int mt = 0; mt < 2; ++mt) {
        v16h a0 = loadA(in, HID, mt * 16 + l16, half, 0);
        v16h a1 = loadA(in, HID, mt * 16 + l16, half, 32);
        v8f acc = {0.f, 0.f, 0.f, 0.f, 0.f, 0.f, 0.f, 0.f};
        acc = wmma_f16(a0, loadB(W3t, HID, l16, half, 0), acc);
        acc = wmma_f16(a1, loadB(W3t, HID, l16, half, 32), acc);
        if (l16 < ncols) {
            float bv = bias[l16];
#pragma unroll
            for (int r = 0; r < 8; ++r) {
                int m = mt * 16 + r + 8 * half;
                outTile[m * outLd + l16] = acc[r] + bv;
            }
        }
    }
}

__global__ __launch_bounds__(128) void ppo_cbf_fused(
    const float* __restrict__ x, const float* __restrict__ sdfs,
    const float* __restrict__ grads, const float* __restrict__ f_x,
    const float* __restrict__ g_x, const float* __restrict__ action,
    const float* __restrict__ cW1, const float* __restrict__ cb1,
    const float* __restrict__ cW2, const float* __restrict__ cb2,
    const float* __restrict__ cW3, const float* __restrict__ cb3,
    const float* __restrict__ aW1, const float* __restrict__ ab1,
    const float* __restrict__ aW2, const float* __restrict__ ab2,
    const float* __restrict__ aW3, const float* __restrict__ ab3,
    const float* __restrict__ logstd, const float* __restrict__ Lmat,
    const float* __restrict__ s0,
    float* __restrict__ out_u, float* __restrict__ out_lp,
    float* __restrict__ out_ent, float* __restrict__ out_val, int N) {
    // ---- LDS ----
    __shared__ __align__(16) _Float16 sW1a[HID * 32];   // transposed, K padded 17->32
    __shared__ __align__(16) _Float16 sW2a[HID * HID];  // transposed
    __shared__ __align__(16) _Float16 sW3a[16 * HID];   // transposed, N padded 4->16
    __shared__ __align__(16) _Float16 sW1c[HID * 32];
    __shared__ __align__(16) _Float16 sW2c[HID * HID];
    __shared__ __align__(16) _Float16 sW3c[16 * HID];
    __shared__ float sB1a[HID], sB2a[HID], sB3a[ACT];
    __shared__ float sB1c[HID], sB2c[HID], sB3c[1];
    __shared__ __align__(16) _Float16 sX[WAVES][32 * 32];   // x padded 17->32
    __shared__ __align__(16) _Float16 sH[WAVES][32 * HID];  // activations (reused)
    __shared__ float sMean[WAVES][32][ACT];
    __shared__ float sVal[WAVES][32];

    const int tid = threadIdx.x;
    const int wave = tid >> 5, lane = tid & 31;

    // ---- stage weights (transposed, f32 -> f16) ----
    for (int i = tid; i < HID * 32; i += 128) {
        int n = i >> 5, k = i & 31;
        sW1a[i] = (k < OBS) ? (_Float16)aW1[k * HID + n] : (_Float16)0.f;
        sW1c[i] = (k < OBS) ? (_Float16)cW1[k * HID + n] : (_Float16)0.f;
    }
    for (int i = tid; i < HID * HID; i += 128) {
        int n = i >> 6, k = i & 63;
        sW2a[i] = (_Float16)aW2[k * HID + n];
        sW2c[i] = (_Float16)cW2[k * HID + n];
    }
    for (int i = tid; i < 16 * HID; i += 128) {
        int n = i >> 6, k = i & 63;
        sW3a[i] = (n < ACT) ? (_Float16)aW3[k * ACT + n] : (_Float16)0.f;
        sW3c[i] = (n < 1) ? (_Float16)cW3[k] : (_Float16)0.f;
    }
    if (tid < HID) {
        sB1a[tid] = ab1[tid]; sB2a[tid] = ab2[tid];
        sB1c[tid] = cb1[tid]; sB2c[tid] = cb2[tid];
    }
    if (tid < ACT) sB3a[tid] = ab3[tid];
    if (tid == 0) sB3c[0] = cb3[0];

    // ---- stage x tile (32 samples per wave, K padded with zeros) ----
    const long long srow = (long long)blockIdx.x * ROWS_PER_BLOCK + wave * 32 + lane;
    const bool valid = srow < (long long)N;
    _Float16* myX = sX[wave];
#pragma unroll
    for (int k = 0; k < 32; ++k) {
        float xv = (valid && k < OBS) ? x[srow * OBS + k] : 0.f;
        myX[lane * 32 + k] = (_Float16)xv;
    }
    __syncthreads();

    // ---- actor MLP ----
    _Float16* myH = sH[wave];
    hidden_layer<32>(myX, 32, sW1a, sB1a, myH, lane);
    hidden_layer<64>(myH, HID, sW2a, sB2a, myH, lane);  // in-place: per-wave DS in-order
    final_layer(myH, sW3a, &sMean[wave][0][0], ACT, ACT, sB3a, lane);

    // ---- critic MLP ----
    hidden_layer<32>(myX, 32, sW1c, sB1c, myH, lane);
    hidden_layer<64>(myH, HID, sW2c, sB2c, myH, lane);
    final_layer(myH, sW3c, &sVal[wave][0], 1, 1, sB3c, lane);
    __syncthreads();

    // ---- per-sample scalar tail: CBF-QP + log-prob + entropy ----
    if (valid) {
        // Q = tril(L) tril(L)^T + eps I ; Cholesky (same for all samples)
        float Lt[4][4];
#pragma unroll
        for (int i = 0; i < 4; ++i)
#pragma unroll
            for (int j = 0; j < 4; ++j)
                Lt[i][j] = (j <= i) ? Lmat[i * 4 + j] : 0.f;
        float Q[4][4];
#pragma unroll
        for (int i = 0; i < 4; ++i)
#pragma unroll
            for (int j = 0; j < 4; ++j) {
                float s = 0.f;
#pragma unroll
                for (int k = 0; k < 4; ++k) s += Lt[i][k] * Lt[j][k];
                Q[i][j] = s + ((i == j) ? EPSQ : 0.f);
            }
        float C[4][4];
#pragma unroll
        for (int j = 0; j < 4; ++j) {
            float d = Q[j][j];
#pragma unroll
            for (int k = 0; k < 4; ++k)
                if (k < j) d -= C[j][k] * C[j][k];
            C[j][j] = sqrtf(d);
#pragma unroll
            for (int i = 0; i < 4; ++i) {
                if (i > j) {
                    float s = Q[i][j];
#pragma unroll
                    for (int k = 0; k < 4; ++k)
                        if (k < j) s -= C[i][k] * C[j][k];
                    C[i][j] = s / C[j][j];
                }
            }
        }

        float g[OBS], fx[OBS];
#pragma unroll
        for (int s = 0; s < OBS; ++s) {
            g[s] = grads[srow * OBS + s];
            fx[s] = f_x[srow * OBS + s];
        }
        float G[4] = {0.f, 0.f, 0.f, 0.f};
        float gf = 0.f;
#pragma unroll
        for (int s = 0; s < OBS; ++s) {
            float gs = g[s];
            gf += gs * fx[s];
            const float* gx = g_x + (srow * OBS + s) * ACT;
#pragma unroll
            for (int a = 0; a < 4; ++a) G[a] -= gs * gx[a];  // G = -grads . g_x
        }
        float act[4];
#pragma unroll
        for (int a = 0; a < 4; ++a) act[a] = action[srow * 4 + a];
        // h = s0 + sdf + grads.f_x + grads.g_x.action = s0 + sdf + gf - G.act
        float hq = s0[0] + sdfs[srow] + gf -
                   (G[0] * act[0] + G[1] * act[1] + G[2] * act[2] + G[3] * act[3]);

        // v = Q^{-1} G via two triangular solves
        float y0 = G[0] / C[0][0];
        float y1 = (G[1] - C[1][0] * y0) / C[1][1];
        float y2 = (G[2] - C[2][0] * y0 - C[2][1] * y1) / C[2][2];
        float y3 = (G[3] - C[3][0] * y0 - C[3][1] * y1 - C[3][2] * y2) / C[3][3];
        float v3 = y3 / C[3][3];
        float v2 = (y2 - C[3][2] * v3) / C[2][2];
        float v1 = (y1 - C[2][1] * v2 - C[3][1] * v3) / C[1][1];
        float v0 = (y0 - C[1][0] * v1 - C[2][0] * v2 - C[3][0] * v3) / C[0][0];
        float vv[4] = {v0, v1, v2, v3};
        float denom = G[0] * v0 + G[1] * v1 + G[2] * v2 + G[3] * v3;
        float lam = fmaxf(0.f, -hq / denom);

        float lp = 0.f, ent = 0.f;
#pragma unroll
        for (int a = 0; a < 4; ++a) {
            float mean = sMean[wave][lane][a];
            float ls = logstd[a];
            float z = (act[a] - mean) * __expf(-ls);
            lp += -0.5f * z * z - ls - 0.5f * LOG2PI;
            ent += 0.5f + 0.5f * LOG2PI + ls;
            out_u[srow * 4 + a] = mean - lam * vv[a];
        }
        out_lp[srow] = lp;
        out_ent[srow] = ent;
        out_val[srow] = sVal[wave][lane];
    }
}

extern "C" void kernel_launch(void* const* d_in, const int* in_sizes, int n_in,
                              void* d_out, int out_size, void* d_ws, size_t ws_size,
                              hipStream_t stream) {
    const float* x      = (const float*)d_in[0];
    const float* sdfs   = (const float*)d_in[1];
    const float* grads  = (const float*)d_in[2];
    const float* f_x    = (const float*)d_in[3];
    const float* g_x    = (const float*)d_in[4];
    const float* action = (const float*)d_in[5];
    const float* cW1 = (const float*)d_in[6];
    const float* cb1 = (const float*)d_in[7];
    const float* cW2 = (const float*)d_in[8];
    const float* cb2 = (const float*)d_in[9];
    const float* cW3 = (const float*)d_in[10];
    const float* cb3 = (const float*)d_in[11];
    const float* aW1 = (const float*)d_in[12];
    const float* ab1 = (const float*)d_in[13];
    const float* aW2 = (const float*)d_in[14];
    const float* ab2 = (const float*)d_in[15];
    const float* aW3 = (const float*)d_in[16];
    const float* ab3 = (const float*)d_in[17];
    const float* logstd = (const float*)d_in[18];
    const float* Lmat   = (const float*)d_in[19];
    const float* s0     = (const float*)d_in[20];

    const int N = in_sizes[0] / OBS;
    float* out = (float*)d_out;
    float* out_u   = out;
    float* out_lp  = out + (size_t)4 * N;
    float* out_ent = out + (size_t)5 * N;
    float* out_val = out + (size_t)6 * N;

    const int blocks = (N + ROWS_PER_BLOCK - 1) / ROWS_PER_BLOCK;
    hipLaunchKernelGGL(ppo_cbf_fused, dim3(blocks), dim3(128), 0, stream,
                       x, sdfs, grads, f_x, g_x, action,
                       cW1, cb1, cW2, cb2, cW3, cb3,
                       aW1, ab1, aW2, ab2, aW3, ab3,
                       logstd, Lmat, s0,
                       out_u, out_lp, out_ent, out_val, N);
}
